// DummyLinear4bit_90658169684562
// MI455X (gfx1250) — compile-verified, matrix operand order checked by reading, not run on hardware
//
#include <hip/hip_runtime.h>

typedef _Float16 half_t;
typedef __attribute__((ext_vector_type(16))) _Float16 v16h;
typedef __attribute__((ext_vector_type(8)))  _Float16 v8h;
typedef __attribute__((ext_vector_type(8)))  float    v8f;
typedef __attribute__((ext_vector_type(4)))  int      v4i;

#define OUT_DIM 14336
#define IN_DIM  4096
#define BATCH   32
#define NB_BLOCKS ((OUT_DIM * IN_DIM) / 64)   // 917504 per-64-element blocks

// ---------------------------------------------------------------------------
// Kernel 1: fold the nested double-quant into one (scale, offset*scale) pair
// per 64-element block.  cs = (absmax/code) * (g_absmax/g_code), offc = off*cs
// ---------------------------------------------------------------------------
__global__ __launch_bounds__(256) void nf4_fold_scales(
    const float* __restrict__ absmax,
    const float* __restrict__ code,
    const float* __restrict__ offset,
    const float* __restrict__ g_absmax,
    const float* __restrict__ g_code,
    float* __restrict__ cs_out,
    float* __restrict__ offc_out)
{
    int b = blockIdx.x * blockDim.x + threadIdx.x;
    if (b >= NB_BLOCKS) return;
    int g = b >> 2;                                   // 4 blocks per 256-elt group
    float cs = (absmax[b] / code[b]) * (g_absmax[g] / g_code[g]);
    cs_out[b]   = cs;
    offc_out[b] = offset[b] * cs;
}

// ---------------------------------------------------------------------------
// Kernel 2: streaming NF4-dequant GEMM, out[32,14336] = x[32,4096] @ W^T
// One wave computes a 32x16 output tile (2 x v_wmma_f32_16x16x32_f16 per chunk)
// ---------------------------------------------------------------------------
__global__ __launch_bounds__(256) void nf4_gemm_wmma(
    const half_t* __restrict__ x,        // [32, 4096] fp16
    const int*    __restrict__ packed,   // [N/2] int32, one byte value each
    const float*  __restrict__ cs_tab,   // [NB] folded scale
    const float*  __restrict__ offc_tab, // [NB] folded offset*scale
    float*        __restrict__ out)      // [32, 14336] f32
{
    const int lane  = threadIdx.x & 31;
    const int wave  = threadIdx.x >> 5;
    const int tileN = blockIdx.x * 8 + wave;     // 0..895
    const int o0    = tileN * 16;
    const int n     = lane & 15;                 // column within 16-wide tile
    const int o     = o0 + n;                    // this lane's OUT row of W
    const int khalf = (lane >> 4) * 8;           // ISA 16-bit A/B lane K-split

    const half_t* xrow0 = x + n * IN_DIM;        // A rows 0..15 fragment
    const half_t* xrow1 = x + (n + 16) * IN_DIM; // A rows 16..31 fragment
    const int*    prow  = packed + o * (IN_DIM / 2);

    union AFrag { v16h v; v8h h[2]; };
    union BFrag { v16h v; half_t e[16]; };

    v8f acc0 = {};   // batch rows 0..15
    v8f acc1 = {};   // batch rows 16..31

    #pragma unroll 2
    for (int k0 = 0; k0 < IN_DIM; k0 += 32) {
        const int kb = k0 + khalf;

        // ---- A fragments (16x32 f16, lanes 0-15: K {0-7,16-23}; 16-31: {8-15,24-31})
        AFrag a0, a1;
        a0.h[0] = *(const v8h*)(xrow0 + kb);
        a0.h[1] = *(const v8h*)(xrow0 + kb + 16);
        a1.h[0] = *(const v8h*)(xrow1 + kb);
        a1.h[1] = *(const v8h*)(xrow1 + kb + 16);

        // ---- packed weight bytes: K kb..kb+7 and kb+16..kb+23 (1 byte / int32)
        v4i p0 = *(const v4i*)(prow + (kb >> 1));
        v4i p1 = *(const v4i*)(prow + ((kb + 16) >> 1));

        // keep the HBM weight stream ahead of us
        __builtin_prefetch(prow + ((kb + 512) >> 1), 0, 1);

        // whole lane K-range stays inside one 64-element block (k0 % 32 == 0)
        const int blk  = (o << 6) + (k0 >> 6);   // 64 blocks per OUT row
        const float cs   = cs_tab[blk];
        const float offc = offc_tab[blk];

        // ---- register dequant: even elem = low nibble, odd = high nibble
        BFrag bf;
        #pragma unroll
        for (int j = 0; j < 4; ++j) {
            int byte0 = p0[j];
            bf.e[2 * j + 0] = (half_t)((float)(byte0 & 15)        * cs - offc);
            bf.e[2 * j + 1] = (half_t)((float)((byte0 >> 4) & 15) * cs - offc);
            int byte1 = p1[j];
            bf.e[8 + 2 * j + 0] = (half_t)((float)(byte1 & 15)        * cs - offc);
            bf.e[8 + 2 * j + 1] = (half_t)((float)((byte1 >> 4) & 15) * cs - offc);
        }

        // ---- D = A x B + C  (f32 accumulate)
        acc0 = __builtin_amdgcn_wmma_f32_16x16x32_f16(
            false, a0.v, false, bf.v, (short)0, acc0, false, false);
        acc1 = __builtin_amdgcn_wmma_f32_16x16x32_f16(
            false, a1.v, false, bf.v, (short)0, acc1, false, false);
    }

    // ---- store: 16x16 f32 D layout — lane 0-15: M=vgpr, lane 16-31: M=vgpr+8
    const int mbase = (lane >> 4) * 8;
    float* ocol = out + o0 + n;
    #pragma unroll
    for (int v = 0; v < 8; ++v) {
        ocol[(mbase + v)      * OUT_DIM] = acc0[v];
        ocol[(mbase + v + 16) * OUT_DIM] = acc1[v];
    }
}

// ---------------------------------------------------------------------------
extern "C" void kernel_launch(void* const* d_in, const int* in_sizes, int n_in,
                              void* d_out, int out_size, void* d_ws, size_t ws_size,
                              hipStream_t stream) {
    (void)in_sizes; (void)n_in; (void)out_size; (void)ws_size;

    const half_t* x        = (const half_t*)d_in[0];
    const int*    packed   = (const int*)   d_in[1];
    const float*  absmax   = (const float*) d_in[2];
    const float*  code     = (const float*) d_in[3];
    const float*  offset   = (const float*) d_in[4];
    const float*  g_absmax = (const float*) d_in[5];
    const float*  g_code   = (const float*) d_in[6];
    float*        out      = (float*)d_out;

    float* cs   = (float*)d_ws;          // [NB]
    float* offc = cs + NB_BLOCKS;        // [NB]  (total ws use: 7.3 MB)

    nf4_fold_scales<<<(NB_BLOCKS + 255) / 256, 256, 0, stream>>>(
        absmax, code, offset, g_absmax, g_code, cs, offc);

    nf4_gemm_wmma<<<OUT_DIM / (16 * 8), 256, 0, stream>>>(
        x, packed, cs, offc, out);
}
